// DRNetwork_13176959664128
// MI455X (gfx1250) — matrix operand order repaired and checked by reading, not compiled, not run on hardware
//
#include <hip/hip_runtime.h>
#include <hip/hip_bf16.h>

// ---------------- problem constants (from reference) ----------------
constexpr int BG   = 16;      // graphs
constexpr int NPG  = 2048;    // nodes per graph
constexpr int NN   = BG * NPG;// 32768 total nodes
constexpr int IND  = 32;      // input dim
constexpr int HID  = 128;
constexpr int OUTD = 64;
constexpr int KNN  = 16;      // neighbors
constexpr int NK1  = KNN + 1; // 17
constexpr int PP   = 100000;  // pairs

// ---------------- WMMA types ----------------
typedef __attribute__((ext_vector_type(16))) __bf16        v16bf;
typedef __attribute__((ext_vector_type(8)))  float         v8f;
typedef __attribute__((ext_vector_type(4)))  unsigned int  v4u;

__device__ __forceinline__ unsigned short f2bf(float f) {
  unsigned u = __float_as_uint(f);
  u += 0x7fffu + ((u >> 16) & 1u);      // round-to-nearest-even
  return (unsigned short)(u >> 16);
}

// A fragment: 16x32 bf16, A row-major with row stride lda (elements).
// Lanes 0-15: M=lane, K in {0..15}; lanes 16-31: K in {16..31}. u[e] holds
// k = kh + e.
__device__ __forceinline__ v16bf load_a_frag(const unsigned short* A, int lda,
                                             int m0, int k0, int lane) {
  union { v16bf v; unsigned int d[8]; } f;
  int m  = m0 + (lane & 15);
  int kh = (lane >> 4) << 4;                      // 0 or 16
  const unsigned short* row = A + (size_t)m * lda + (k0 + kh);
#pragma unroll
  for (int i = 0; i < 8; ++i)
    f.d[i] = *(const unsigned int*)(row + 2 * i);           // K = kh + 2i, 2i+1
  return f.v;
}

// B fragment from transposed storage: Bt row-major [N x K] row stride ldt,
// i.e. B(k,n) = Bt[n*ldt + k] -> contiguous dword loads per lane.
// B layout: lane = (n&15) + 16*(k>=16 within tile); u[e] holds k = kb + e.
__device__ __forceinline__ v16bf load_bt_frag(const unsigned short* Bt, int ldt,
                                              int k0, int n0, int lane) {
  union { v16bf v; unsigned int d[8]; } f;
  int n  = n0 + (lane & 15);
  int kb = k0 + ((lane >> 4) << 4);
  const unsigned short* row = Bt + (size_t)n * ldt + kb;
#pragma unroll
  for (int i = 0; i < 8; ++i)
    f.d[i] = *(const unsigned int*)(row + 2 * i);
  return f.v;
}

// B fragment from pre-packed (fragment-major) storage: per 32x16 tile, lane's 16
// bf16 stored contiguously -> two global_load_b128 per fragment.
__device__ __forceinline__ v16bf load_bp_frag(const unsigned short* Bp, int Nd,
                                              int k0, int n0, int lane) {
  union { v16bf v; v4u q[2]; } f;
  const v4u* p = (const v4u*)(Bp +
      ((((size_t)(k0 >> 5) * (size_t)(Nd >> 4)) + (size_t)(n0 >> 4)) * 32 + lane) * 16);
  f.q[0] = p[0];
  f.q[1] = p[1];
  return f.v;
}

__device__ __forceinline__ v8f wmma_bf16(v16bf a, v16bf b, v8f c) {
  return __builtin_amdgcn_wmma_f32_16x16x32_bf16(false, a, false, b, (short)0, c, false, false);
}

// ---------------- kernel: fp32 weights -> bf16 in WMMA B-fragment layout ----------------
// dst[((tk*(N/16)+tn)*32 + lane)*16 + e] = bf16(src[k*N+n]),
//   tk=k/32, tn=n/16, lane=(n&15)|(((k>>4)&1)<<4), e=k&15.
__global__ void k_pack_b(const float* __restrict__ src, unsigned short* __restrict__ dst,
                         int Kd, int Nd) {
  int t = blockIdx.x * blockDim.x + threadIdx.x;
  if (t >= Kd * Nd) return;
  int k = t / Nd, n = t - k * Nd;
  int lane = (n & 15) | (((k >> 4) & 1) << 4);
  size_t di = ((((size_t)(k >> 5)) * (Nd >> 4) + (n >> 4)) * 32 + lane) * 16 + (k & 15);
  dst[di] = f2bf(src[(size_t)k * Nd + n]);
}

// ---------------- kernel: input linear + sq norms ----------------
__global__ __launch_bounds__(HID) void k_linear(const float* __restrict__ x,
                                                const float* __restrict__ W,
                                                const float* __restrict__ b,
                                                float* __restrict__ h,
                                                unsigned short* __restrict__ h16,
                                                float* __restrict__ sq) {
  __shared__ float xs[IND];
  __shared__ float red[HID];
  int n = blockIdx.x, j = threadIdx.x;
  if (j < IND) xs[j] = x[(size_t)n * IND + j];
  __syncthreads();
  float acc = b[j];
#pragma unroll
  for (int k = 0; k < IND; ++k) acc = fmaf(xs[k], W[k * HID + j], acc);
  h  [(size_t)n * HID + j] = acc;
  h16[(size_t)n * HID + j] = f2bf(acc);
  red[j] = acc * acc;
  __syncthreads();
  for (int s = HID / 2; s > 0; s >>= 1) {
    if (j < s) red[j] += red[j + s];
    __syncthreads();
  }
  if (j == 0) sq[n] = red[0];
}

// ---------------- kernel: distance GEMM (WMMA) + top-16 selection ----------------
// One block = one 16-row band of one graph. A band (16x128) held in registers
// (4 fragments). 8 waves compute 16x16 d2 tiles into a 16x2048 LDS band; then
// each wave extracts the 16 smallest entries for 2 rows.
__global__ __launch_bounds__(256) void k_dist_topk(const unsigned short* __restrict__ h16,
                                                   const float* __restrict__ sq,
                                                   int* __restrict__ idx_out) {
  __shared__ float d2s[16 * NPG];           // 128 KB (CDNA5: 320 KB LDS/WGP)
  const int blk  = blockIdx.x;              // 0..2047
  const int g    = blk >> 7;                // graph
  const int band = blk & 127;               // 16-row band within graph
  const int gi0  = g * NPG + band * 16;
  const int tid  = threadIdx.x;
  const int wave = tid >> 5, lane = tid & 31;
  const unsigned short* hg  = h16 + (size_t)g * NPG * HID;
  const float*          sqg = sq + g * NPG;
  const float inf = __builtin_inff();

  // loop-invariant A band fragments + per-lane row norms
  v16bf afr[4];
#pragma unroll
  for (int kk = 0; kk < 4; ++kk)
    afr[kk] = load_a_frag(hg, HID, band * 16, kk * 32, lane);
  const int mb = (lane >> 4) << 3;
  float sqm[8];
#pragma unroll
  for (int r = 0; r < 8; ++r) sqm[r] = sqg[band * 16 + mb + r];

  // ---- compute phase: wave w handles column tiles w, w+8, ... ----
  for (int t = wave; t < NPG / 16; t += 8) {
    int j0 = t * 16;
    v8f acc = {};
#pragma unroll
    for (int kk = 0; kk < 4; ++kk) {
      v16bf b = load_bt_frag(hg, HID, kk * 32, j0, lane);    // B = hg^T
      acc = wmma_bf16(afr[kk], b, acc);
    }
    int n  = j0 + (lane & 15);
    float sn = sqg[n];
#pragma unroll
    for (int r = 0; r < 8; ++r) {
      float d2 = sqm[r] + sn - 2.0f * acc[r];
      if (band * 16 + mb + r == n) d2 = inf;                 // exclude self
      d2s[(mb + r) * NPG + n] = d2;
    }
  }
  __syncthreads();

  // ---- selection phase: wave handles rows 2*wave, 2*wave+1 ----
  for (int rr = 0; rr < 2; ++rr) {
    int row  = wave * 2 + rr;
    int grow = gi0 + row;
    float* drow = d2s + row * NPG;
    for (int s = 0; s < KNN; ++s) {
      float best = inf;
      int   bi   = 0x7fffffff;
      for (int c = lane; c < NPG; c += 32) {
        float v = drow[c];
        if (v < best) { best = v; bi = c; }
      }
#pragma unroll
      for (int off = 16; off > 0; off >>= 1) {
        float ov = __shfl_xor(best, off, 32);
        int   oi = __shfl_xor(bi,   off, 32);
        if (ov < best || (ov == best && oi < bi)) { best = ov; bi = oi; }
      }
      if (lane == 0) {
        idx_out[(size_t)grow * NK1 + s] = g * NPG + bi;
        drow[bi] = inf;                                  // extract
      }
      asm volatile("s_wait_dscnt 0" ::: "memory");       // wave32 lockstep: DS write visible
    }
    if (lane == 0) idx_out[(size_t)grow * NK1 + KNN] = grow;  // self loop
  }
}

// ---------------- kernel: generic bf16 WMMA GEMM, bias + optional ReLU ----------------
// C[M,N] = act(A[M,K] * Bpacked[K,N] + bias). Wave computes 16(M)x64(N) with 4
// accumulators (A-fragment reuse x4). Block = 4 waves = 64x64 of C.
// Grid: (M/64, N/64). B is in fragment-major packed layout (k_pack_b).
__global__ __launch_bounds__(128) void k_gemm(const unsigned short* __restrict__ A,
                                              const unsigned short* __restrict__ Bp,
                                              const float* __restrict__ bias,
                                              unsigned short* __restrict__ out16,
                                              float* __restrict__ out32,
                                              int M, int N, int K, int relu) {
  const int lane = threadIdx.x & 31;
  const int wave = threadIdx.x >> 5;
  const int m0 = blockIdx.x * 64 + wave * 16;
  const int n0 = blockIdx.y * 64;
  v8f acc[4] = {};
  for (int k0 = 0; k0 < K; k0 += 32) {
    v16bf a = load_a_frag(A, K, m0, k0, lane);
#pragma unroll
    for (int t = 0; t < 4; ++t) {
      v16bf b = load_bp_frag(Bp, N, k0, n0 + 16 * t, lane);
      acc[t] = wmma_bf16(a, b, acc[t]);
    }
  }
  const int mb = m0 + ((lane >> 4) << 3);
#pragma unroll
  for (int t = 0; t < 4; ++t) {
    int n  = n0 + 16 * t + (lane & 15);
    float bn = bias ? bias[n] : 0.0f;
#pragma unroll
    for (int r = 0; r < 8; ++r) {
      float v = acc[t][r] + bn;
      if (relu) v = fmaxf(v, 0.0f);
      if (out16) out16[(size_t)(mb + r) * N + n] = f2bf(v);
      if (out32) out32[(size_t)(mb + r) * N + n] = v;
    }
  }
}

// ---------------- kernel: attention projections a_src/a_dst ----------------
__global__ void k_att(const float* __restrict__ xl,
                      const float* __restrict__ ws, const float* __restrict__ wd,
                      float* __restrict__ a_src, float* __restrict__ a_dst) {
  int n = blockIdx.x * blockDim.x + threadIdx.x;
  if (n >= NN) return;
  float s = 0.f, d = 0.f;
  const float* row = xl + (size_t)n * HID;
#pragma unroll 8
  for (int k = 0; k < HID; ++k) {
    float v = row[k];
    s = fmaf(v, ws[k], s);
    d = fmaf(v, wd[k], d);
  }
  a_src[n] = s;
  a_dst[n] = d;
}

// ---------------- kernel: GAT softmax + aggregate ----------------
__global__ __launch_bounds__(HID) void k_gat(const float* __restrict__ xl,
                                             const int* __restrict__ idx,
                                             const float* __restrict__ a_src,
                                             const float* __restrict__ a_dst,
                                             const float* __restrict__ gat_b,
                                             unsigned short* __restrict__ gout16) {
  __shared__ float lg[NK1];
  __shared__ float wsm[NK1];
  __shared__ int   nb[NK1];
  __shared__ float invs;
  const int n = blockIdx.x, j = threadIdx.x;
  if (j < NK1) {
    int id = idx[(size_t)n * NK1 + j];
    nb[j] = id;
    float z = a_src[id] + a_dst[n];
    lg[j] = (z > 0.f) ? z : 0.2f * z;                // leaky_relu(0.2)
  }
  __syncthreads();
  if (j < NK1) {                                     // 17 exps per node (cooperative)
    float mx = -__builtin_inff();
#pragma unroll
    for (int k = 0; k < NK1; ++k) mx = fmaxf(mx, lg[k]);
    wsm[j] = __expf(lg[j] - mx);
  }
  __syncthreads();
  if (j == 0) {
    float s = 0.f;
#pragma unroll
    for (int k = 0; k < NK1; ++k) s += wsm[k];
    invs = 1.0f / s;
  }
  __syncthreads();
  const float inv = invs;
  float acc = gat_b[j];
#pragma unroll
  for (int k = 0; k < NK1; ++k)
    acc = fmaf(wsm[k] * inv, xl[(size_t)nb[k] * HID + j], acc);
  gout16[(size_t)n * HID + j] = f2bf(acc);
}

// ---------------- kernel: pair gather + labels ----------------
__global__ void k_pairs(const float* __restrict__ emb,
                        const int* __restrict__ pairs,
                        const float* __restrict__ labels,
                        float* __restrict__ out) {
  const int total = 2 * PP * OUTD;
  int t = blockIdx.x * blockDim.x + threadIdx.x;
  if (t < total) {
    int s   = t / (PP * OUTD);
    int rem = t - s * (PP * OUTD);
    int p   = rem >> 6;            // /OUTD
    int j   = rem & (OUTD - 1);
    int node = pairs[p * 2 + s];
    out[t] = emb[(size_t)node * OUTD + j];
  } else if (t < total + PP) {
    out[t] = labels[t - total];
  }
}

// ---------------- host launch ----------------
static inline size_t align256(size_t x) { return (x + 255) & ~(size_t)255; }

extern "C" void kernel_launch(void* const* d_in, const int* in_sizes, int n_in,
                              void* d_out, int out_size, void* d_ws, size_t ws_size,
                              hipStream_t stream) {
  (void)in_sizes; (void)n_in; (void)out_size; (void)ws_size;
  const float* x      = (const float*)d_in[0];
  /* d_in[1] = batch (unused) */
  const int*   pairs  = (const int*)  d_in[2];
  const float* labels = (const float*)d_in[3];
  const float* linW   = (const float*)d_in[4];
  const float* linb   = (const float*)d_in[5];
  const float* gatW   = (const float*)d_in[6];
  const float* attS   = (const float*)d_in[7];
  const float* attD   = (const float*)d_in[8];
  const float* gatb   = (const float*)d_in[9];
  const float* W1     = (const float*)d_in[10];
  const float* b1     = (const float*)d_in[11];
  const float* W2     = (const float*)d_in[12];
  const float* b2     = (const float*)d_in[13];
  const float* W3     = (const float*)d_in[14];
  const float* b3     = (const float*)d_in[15];
  float* out = (float*)d_out;

  // workspace carve
  char* ws = (char*)d_ws;
  size_t off = 0;
  auto alloc = [&](size_t bytes) { char* p = ws + off; off += align256(bytes); return p; };
  float*          h     = (float*)         alloc((size_t)NN * HID * 4);
  unsigned short* h16   = (unsigned short*)alloc((size_t)NN * HID * 2);
  float*          sq    = (float*)         alloc((size_t)NN * 4);
  int*            idx   = (int*)           alloc((size_t)NN * NK1 * 4);
  float*          xl    = (float*)         alloc((size_t)NN * HID * 4);
  float*          asrc  = (float*)         alloc((size_t)NN * 4);
  float*          adst  = (float*)         alloc((size_t)NN * 4);
  unsigned short* g16   = (unsigned short*)alloc((size_t)NN * HID * 2);
  unsigned short* a1    = (unsigned short*)alloc((size_t)NN * 4 * HID * 2);
  unsigned short* a2    = (unsigned short*)alloc((size_t)NN * 2 * HID * 2);
  float*          emb   = (float*)         alloc((size_t)NN * OUTD * 4);
  unsigned short* wgat  = (unsigned short*)alloc((size_t)HID * HID * 2);
  unsigned short* w1b   = (unsigned short*)alloc((size_t)HID * 4 * HID * 2);
  unsigned short* w2b   = (unsigned short*)alloc((size_t)4 * HID * 2 * HID * 2);
  unsigned short* w3b   = (unsigned short*)alloc((size_t)2 * HID * OUTD * 2);

  // 1) pack weights into WMMA B-fragment layout (bf16)
  {
    int n;
    n = HID * HID;
    k_pack_b<<<(n + 255) / 256, 256, 0, stream>>>(gatW, wgat, HID, HID);
    n = HID * 4 * HID;
    k_pack_b<<<(n + 255) / 256, 256, 0, stream>>>(W1, w1b, HID, 4 * HID);
    n = 4 * HID * 2 * HID;
    k_pack_b<<<(n + 255) / 256, 256, 0, stream>>>(W2, w2b, 4 * HID, 2 * HID);
    n = 2 * HID * OUTD;
    k_pack_b<<<(n + 255) / 256, 256, 0, stream>>>(W3, w3b, 2 * HID, OUTD);
  }
  // 2) input linear
  k_linear<<<NN, HID, 0, stream>>>(x, linW, linb, h, h16, sq);
  // 3) kNN via WMMA distance + top-16
  k_dist_topk<<<NN / 16, 256, 0, stream>>>(h16, sq, idx);
  // 4) GAT linear: xl = h @ gat_W (no bias)
  k_gemm<<<dim3(NN / 64, HID / 64), 128, 0, stream>>>(
      h16, wgat, nullptr, nullptr, xl, NN, HID, HID, 0);
  // 5) attention projections
  k_att<<<(NN + 255) / 256, 256, 0, stream>>>(xl, attS, attD, asrc, adst);
  // 6) softmax-aggregate
  k_gat<<<NN, HID, 0, stream>>>(xl, idx, asrc, adst, gatb, g16);
  // 7) MLP
  k_gemm<<<dim3(NN / 64, (4 * HID) / 64), 128, 0, stream>>>(
      g16, w1b, b1, a1, nullptr, NN, 4 * HID, HID, 1);
  k_gemm<<<dim3(NN / 64, (2 * HID) / 64), 128, 0, stream>>>(
      a1, w2b, b2, a2, nullptr, NN, 2 * HID, 4 * HID, 1);
  k_gemm<<<dim3(NN / 64, OUTD / 64), 128, 0, stream>>>(
      a2, w3b, b3, nullptr, emb, NN, OUTD, 2 * HID, 0);
  // 8) pair gather + labels
  {
    int total = 2 * PP * OUTD + PP;
    k_pairs<<<(total + 255) / 256, 256, 0, stream>>>(emb, pairs, labels, out);
  }
}
